// customGraphSAGE3_29154238005822
// MI455X (gfx1250) — compile-verified, matrix operand order checked by reading, not run on hardware
//
#include <hip/hip_runtime.h>
#include <hip/hip_bf16.h>

typedef __attribute__((ext_vector_type(16))) _Float16 v16h;
typedef __attribute__((ext_vector_type(8)))  float    v8f;
typedef __attribute__((ext_vector_type(4)))  int      v4i;

#define NN 256   // nodes
#define DD 64    // node feature dim
#define EDIM 32  // edge feature dim
#define EE 2048  // edges

// ---- workspace layout (float units) ----
#define WS_AGG    0                       // 256*64
#define WS_DEG    (WS_AGG + NN*DD)        // 256
#define WS_H      (WS_DEG + NN)           // 256*64
#define WS_ZPART  (WS_H + NN*DD)          // 256*256
#define WS_EPART  (WS_ZPART + NN*256)     // 2048*256
#define WS_PRED1  (WS_EPART + EE*256)     // 2048
#define WS_W1T    (WS_PRED1 + EE)         // 128*256 halves, k-slice-major = 16384 float slots
#define WS_W2T    (WS_W1T + 16384)        // 64*128 halves, k-slice-major  = 4096 float slots
// total ~645K floats ~ 2.6 MB

#define AS1 __attribute__((address_space(1)))
#define AS3 __attribute__((address_space(3)))

__device__ __forceinline__ int kmap(int j, int hi) {
    // 16-bit A/B fragment K mapping: VGPR j holds K pair {k0, k0+1}
    return ((j < 4) ? 0 : 16) + hi * 8 + 2 * (j & 3);
}

// 16-byte Global -> LDS copy, async DMA path when available.
__device__ __forceinline__ void async_copy16(const _Float16* g, _Float16* l) {
#if __has_builtin(__builtin_amdgcn_global_load_async_to_lds_b128)
    __builtin_amdgcn_global_load_async_to_lds_b128(
        (AS1 v4i*)g, (AS3 v4i*)l, 0, 0);
#else
    *(float4*)l = *(const float4*)g;   // fallback: through-register copy
#endif
}

__device__ __forceinline__ void wait_async0() {
#if __has_builtin(__builtin_amdgcn_s_wait_asynccnt)
    __builtin_amdgcn_s_wait_asynccnt(0);
#else
    asm volatile("s_wait_asynccnt 0" ::: "memory");
#endif
}

// ---------------- kernel 0: zero agg+deg ----------------
__global__ __launch_bounds__(256) void k_zero(float* ws) {
    int i = blockIdx.x * 256 + threadIdx.x;
    if (i < NN * DD + NN) ws[WS_AGG + i] = 0.0f;
}

// ---------------- kernel 1: msg + segment scatter-add ----------------
__global__ __launch_bounds__(64) void k_msg(const float* __restrict__ x,
                                            const int* __restrict__ ei,
                                            const float* __restrict__ ef,
                                            const float* __restrict__ Wm,
                                            const float* __restrict__ bm,
                                            float* agg, float* deg) {
    __shared__ float in_s[96];
    int e = blockIdx.x, t = threadIdx.x;
    int v = ei[e], u = ei[EE + e];
    in_s[t] = x[v * DD + t];
    if (t < EDIM) in_s[DD + t] = ef[e * EDIM + t];
    __syncthreads();
    float acc = bm[t];
    for (int k = 0; k < DD + EDIM; ++k) acc += in_s[k] * Wm[k * DD + t];
    acc = fmaxf(acc, 0.0f);
    atomicAdd(&agg[u * DD + t], acc);
    if (t == 0) atomicAdd(&deg[u], 1.0f);
}

// ---------------- kernel 2: node update h ----------------
__global__ __launch_bounds__(64) void k_hupd(const float* __restrict__ x,
                                             const float* __restrict__ agg,
                                             const float* __restrict__ deg,
                                             const float* __restrict__ Wu,
                                             const float* __restrict__ bu,
                                             float* h) {
    __shared__ float in_s[128];
    int n = blockIdx.x, t = threadIdx.x;
    float dg = fmaxf(deg[n], 1.0f);
    in_s[t] = x[n * DD + t];
    in_s[DD + t] = agg[n * DD + t] / dg;
    __syncthreads();
    float acc = bu[t];
    for (int k = 0; k < 2 * DD; ++k) acc += in_s[k] * Wu[k * DD + t];
    h[n * DD + t] = fmaxf(acc, 0.0f);
}

// ---------------- kernel 3: z_part = h @ Wz ----------------
__global__ __launch_bounds__(256) void k_zpart(const float* __restrict__ h,
                                               const float* __restrict__ W0, // 224x256
                                               float* zp) {
    __shared__ float hz[DD];
    int z = blockIdx.x, c = threadIdx.x;
    if (c < DD) hz[c] = h[z * DD + c];
    __syncthreads();
    float acc = 0.0f;
    for (int k = 0; k < DD; ++k) acc += hz[k] * W0[(128 + k) * 256 + c];
    zp[z * 256 + c] = acc;
}

// ---------------- kernel 4: edge_part ----------------
__global__ __launch_bounds__(256) void k_epart(const float* __restrict__ h,
                                               const int* __restrict__ ei,
                                               const float* __restrict__ ef,
                                               const float* __restrict__ W0, // 224x256
                                               const float* __restrict__ b0,
                                               float* ep) {
    __shared__ float hv[DD], hu[DD], efs[EDIM];
    int e = blockIdx.x, t = threadIdx.x;
    int v = ei[e], u = ei[EE + e];
    if (t < 64) hv[t] = h[v * DD + t];
    else if (t < 128) hu[t - 64] = h[u * DD + (t - 64)];
    else if (t < 160) efs[t - 128] = ef[e * EDIM + (t - 128)];
    __syncthreads();
    float acc = b0[t];
    for (int k = 0; k < DD; ++k)
        acc += hv[k] * W0[k * 256 + t] + hu[k] * W0[(64 + k) * 256 + t];
    for (int k = 0; k < EDIM; ++k)
        acc += efs[k] * W0[(192 + k) * 256 + t];
    ep[e * 256 + t] = acc;
}

// ---------------- kernel 5: weights -> f16, k-slice-major for async staging ----
// W1Ts[(k/32)][n(128)][k%32], W2Ts[(k/32)][n(64)][k%32]
__global__ __launch_bounds__(256) void k_wconv(const float* __restrict__ W1, // 256x128
                                               const float* __restrict__ W2, // 128x64
                                               _Float16* W1Ts, _Float16* W2Ts) {
    int i = blockIdx.x * 256 + threadIdx.x;
    if (i < 128 * 256) {
        int n = i / 256, k = i % 256;
        W1Ts[(k >> 5) * (128 * 32) + n * 32 + (k & 31)] = (_Float16)W1[k * 128 + n];
    } else {
        int j = i - 128 * 256;
        if (j < 64 * 128) {
            int n = j / 128, k = j % 128;
            W2Ts[(k >> 5) * (64 * 32) + n * 32 + (k & 31)] = (_Float16)W2[k * 64 + n];
        }
    }
}

// ---------------- kernel 6: pred1 = MLP1(concat(h[v],h[u],ef)) ----------------
__global__ __launch_bounds__(256) void k_pred1(const float* __restrict__ h,
                                               const int* __restrict__ ei,
                                               const float* __restrict__ ef,
                                               const float* W0, const float* b0,
                                               const float* W1, const float* b1,
                                               const float* W2, const float* b2,
                                               const float* W3, const float* b3,
                                               float* pred1) {
    __shared__ float in_s[160];
    __shared__ float bufA[256];
    __shared__ float bufB[128];
    int e = blockIdx.x, t = threadIdx.x;
    int v = ei[e], u = ei[EE + e];
    if (t < 64) in_s[t] = h[v * DD + t];
    else if (t < 128) in_s[t] = h[u * DD + (t - 64)];
    else if (t < 160) in_s[t] = ef[e * EDIM + (t - 128)];
    __syncthreads();
    float a1 = b0[t];
    for (int k = 0; k < 160; ++k) a1 += in_s[k] * W0[k * 256 + t];
    bufA[t] = fmaxf(a1, 0.0f);
    __syncthreads();
    float a2 = 0.0f;
    if (t < 128) {
        a2 = b1[t];
        for (int k = 0; k < 256; ++k) a2 += bufA[k] * W1[k * 128 + t];
        bufB[t] = fmaxf(a2, 0.0f);
    }
    __syncthreads();
    if (t < 64) {
        float a3 = b2[t];
        for (int k = 0; k < 128; ++k) a3 += bufB[k] * W2[k * 64 + t];
        bufA[t] = fmaxf(a3, 0.0f);
    }
    __syncthreads();
    if (t == 0) {
        float d = b3[0];
        for (int k = 0; k < 64; ++k) d += bufA[k] * W3[k];
        pred1[e] = d;
    }
}

// ---------------- kernel 7: fused pairwise MLP via WMMA (dominant, ~43 GFLOP) ----------------
// One block per edge, 8 waves, 2 passes of 128 z-rows (16 rows/wave).
// W1 k-slices double-buffered in LDS via async DMA, prefetched one slice ahead so
// the DMA overlaps the 8 WMMAs + A-fragment build of the current slice. W2 staged
// in LDS once. A fragments built in registers from ep+zp with relu + f16 convert.
__global__ __launch_bounds__(256) void k_pairmlp(const int* __restrict__ ei,
                                                 const float* __restrict__ ep,
                                                 const float* __restrict__ zp,
                                                 const _Float16* __restrict__ W1Ts, // [8][128][32]
                                                 const float* __restrict__ b1,
                                                 const _Float16* __restrict__ W2Ts, // [4][64][32]
                                                 const float* __restrict__ b2,
                                                 const float* __restrict__ W3,      // 64
                                                 const float* __restrict__ b3,
                                                 const float* __restrict__ pred1,
                                                 float* __restrict__ out) {
    __shared__ float ep_s[256];
    __shared__ __align__(16) unsigned char buf[8 * 16 * 128 * sizeof(_Float16)]; // 32KB reuse
    __shared__ __align__(16) _Float16 w1s[2][128 * 32];   // 2 x 8KB: W1 slice double buffer
    __shared__ __align__(16) _Float16 w2s[4 * 64 * 32];   // 16KB: all of W2
    __shared__ float w3_s[64];
    __shared__ float ssum_s;
    __shared__ int cnt_s;
    _Float16* l1 = (_Float16*)buf;   // [128 rows][128] f16 (layer1 out)
    float*    l2 = (float*)buf;      // [128 rows][64]  f32 (layer2 out, same bytes)

    const int e = blockIdx.x;
    const int tid = threadIdx.x;
    const int w = tid >> 5;
    const int lane = tid & 31;
    const int lane16 = lane & 15;
    const int hi = lane >> 4;

    const int v = ei[e];
    const int u = ei[EE + e];

    // stage all of W2 (16KB) once: 1024 x 16B chunks, 4 per thread
#pragma unroll
    for (int i = 0; i < 4; ++i) {
        int c = tid + 256 * i;
        async_copy16(W2Ts + c * 8, w2s + c * 8);
    }

    ep_s[tid] = ep[e * 256 + tid];
    if (tid < 64) w3_s[tid] = W3[tid];
    if (tid == 0) { ssum_s = 0.0f; cnt_s = 0; }
    __syncthreads();

    for (int p = 0; p < 2; ++p) {
        const int zrow = p * 128 + w * 16 + lane16;
        const float* zpr = zp + zrow * 256;

        // ---- layer 1: K=256 in 8 steps of 32, N=128 in 8 tiles of 16 ----
        v8f acc[8] = {};

        // prologue: issue slice 0 into buffer 0 (buf0 not read since last pass)
        async_copy16(W1Ts + tid * 8,         &w1s[0][tid * 8]);
        async_copy16(W1Ts + (tid + 256) * 8, &w1s[0][(tid + 256) * 8]);

        for (int kk = 0; kk < 8; ++kk) {
            const int kbase = kk * 32;

            wait_async0();     // slice kk DMA complete (and w2s on first iteration)
            __syncthreads();   // resident + all waves done reading the other buffer

            if (kk < 7) {      // prefetch slice kk+1 into the other buffer
                const _Float16* gsl = W1Ts + (kk + 1) * (128 * 32);
                async_copy16(gsl + tid * 8,         &w1s[(kk + 1) & 1][tid * 8]);
                async_copy16(gsl + (tid + 256) * 8, &w1s[(kk + 1) & 1][(tid + 256) * 8]);
            }
            const _Float16* w1cur = w1s[kk & 1];

            v16h a;
#pragma unroll
            for (int j = 0; j < 8; ++j) {
                int k0 = kbase + kmap(j, hi);
                float x0 = ep_s[k0] + zpr[k0];
                float x1 = ep_s[k0 + 1] + zpr[k0 + 1];
                a[2 * j]     = (_Float16)(x0 > 0.0f ? x0 : 0.0f);
                a[2 * j + 1] = (_Float16)(x1 > 0.0f ? x1 : 0.0f);
            }
#pragma unroll
            for (int nt = 0; nt < 8; ++nt) {
                const _Float16* wr = w1cur + (nt * 16 + lane16) * 32;
                v16h b;
#pragma unroll
                for (int j = 0; j < 8; ++j) {
                    int k0 = kmap(j, hi);
                    b[2 * j]     = wr[k0];
                    b[2 * j + 1] = wr[k0 + 1];
                }
                acc[nt] = __builtin_amdgcn_wmma_f32_16x16x32_f16(
                    false, a, false, b, (short)0, acc[nt], false, false);
            }
        }
        // bias + relu -> LDS f16 (restripe C-layout -> row-major)
        __syncthreads();
#pragma unroll
        for (int nt = 0; nt < 8; ++nt) {
            float bb = b1[nt * 16 + lane16];
#pragma unroll
            for (int r = 0; r < 8; ++r) {
                float xv = acc[nt][r] + bb;
                xv = xv > 0.0f ? xv : 0.0f;
                int row = w * 16 + r + 8 * hi;
                l1[row * 128 + nt * 16 + lane16] = (_Float16)xv;
            }
        }
        __syncthreads();

        // ---- layer 2: K=128 in 4 steps, N=64 in 4 tiles; B from LDS (w2s) ----
        v8f acc2[4] = {};
        const _Float16* ar = l1 + (w * 16 + lane16) * 128;
        for (int kk = 0; kk < 4; ++kk) {
            const int kbase = kk * 32;
            v16h a;
#pragma unroll
            for (int j = 0; j < 8; ++j) {
                int k0 = kbase + kmap(j, hi);
                a[2 * j]     = ar[k0];
                a[2 * j + 1] = ar[k0 + 1];
            }
#pragma unroll
            for (int nt = 0; nt < 4; ++nt) {
                const _Float16* wr = w2s + kk * (64 * 32) + (nt * 16 + lane16) * 32;
                v16h b;
#pragma unroll
                for (int j = 0; j < 8; ++j) {
                    int k0 = kmap(j, hi);
                    b[2 * j]     = wr[k0];
                    b[2 * j + 1] = wr[k0 + 1];
                }
                acc2[nt] = __builtin_amdgcn_wmma_f32_16x16x32_f16(
                    false, a, false, b, (short)0, acc2[nt], false, false);
            }
        }
        __syncthreads(); // all l1 reads done before overwriting bytes as l2
#pragma unroll
        for (int nt = 0; nt < 4; ++nt) {
            float bb = b2[nt * 16 + lane16];
#pragma unroll
            for (int r = 0; r < 8; ++r) {
                float xv = acc2[nt][r] + bb;
                xv = xv > 0.0f ? xv : 0.0f;
                int row = w * 16 + r + 8 * hi;
                l2[row * 64 + nt * 16 + lane16] = xv;
            }
        }
        __syncthreads();

        // ---- layer 3: 64 -> 1 (no relu) + mask + reduce ----
        if (tid < 128) {
            int z2 = p * 128 + tid;
            float dot = b3[0];
            const float* rp = l2 + tid * 64;
#pragma unroll 8
            for (int c = 0; c < 64; ++c) dot += rp[c] * w3_s[c];
            if (z2 != v && z2 != u) {
                atomicAdd(&ssum_s, dot);
                atomicAdd(&cnt_s, 1);
            }
        }
        __syncthreads(); // l2 reads done before next pass reuses buf
    }

    if (tid == 0) {
        out[e] = (pred1[e] + ssum_s) / ((float)cnt_s + 1.0f);
    }
}

// ---------------- launcher ----------------
extern "C" void kernel_launch(void* const* d_in, const int* in_sizes, int n_in,
                              void* d_out, int out_size, void* d_ws, size_t ws_size,
                              hipStream_t stream) {
    // flat leaf order of setup_inputs():
    const float* x      = (const float*)d_in[0];   // 256*64
    const int*   ei     = (const int*)d_in[1];     // 2*2048
    const float* ef     = (const float*)d_in[2];   // 2048*32
    /* d_in[3] = num_nodes (scalar, =256) */
    const float* convWm = (const float*)d_in[4];   // 96x64
    const float* convbm = (const float*)d_in[5];   // 64
    const float* convWu = (const float*)d_in[6];   // 128x64
    const float* convbu = (const float*)d_in[7];   // 64
    const float* m1W0 = (const float*)d_in[8];     // 160x256
    const float* m1b0 = (const float*)d_in[9];
    const float* m1W1 = (const float*)d_in[10];    // 256x128
    const float* m1b1 = (const float*)d_in[11];
    const float* m1W2 = (const float*)d_in[12];    // 128x64
    const float* m1b2 = (const float*)d_in[13];
    const float* m1W3 = (const float*)d_in[14];    // 64x1
    const float* m1b3 = (const float*)d_in[15];
    const float* m2W0 = (const float*)d_in[16];    // 224x256
    const float* m2b0 = (const float*)d_in[17];
    const float* m2W1 = (const float*)d_in[18];    // 256x128
    const float* m2b1 = (const float*)d_in[19];
    const float* m2W2 = (const float*)d_in[20];    // 128x64
    const float* m2b2 = (const float*)d_in[21];
    const float* m2W3 = (const float*)d_in[22];    // 64x1
    const float* m2b3 = (const float*)d_in[23];

    float* ws = (float*)d_ws;
    float* agg   = ws + WS_AGG;
    float* deg   = ws + WS_DEG;
    float* h     = ws + WS_H;
    float* zpart = ws + WS_ZPART;
    float* epart = ws + WS_EPART;
    float* pred1 = ws + WS_PRED1;
    _Float16* W1Ts = (_Float16*)(ws + WS_W1T);
    _Float16* W2Ts = (_Float16*)(ws + WS_W2T);
    float* out = (float*)d_out;

    k_zero<<<(NN * DD + NN + 255) / 256, 256, 0, stream>>>(ws);
    k_msg<<<EE, 64, 0, stream>>>(x, ei, ef, convWm, convbm, agg, deg);
    k_hupd<<<NN, 64, 0, stream>>>(x, agg, deg, convWu, convbu, h);
    k_zpart<<<NN, 256, 0, stream>>>(h, m2W0, zpart);
    k_epart<<<EE, 256, 0, stream>>>(h, ei, ef, m2W0, m2b0, epart);
    k_wconv<<<(128 * 256 + 64 * 128 + 255) / 256, 256, 0, stream>>>(m2W1, m2W2, W1Ts, W2Ts);
    k_pred1<<<EE, 256, 0, stream>>>(h, ei, ef, m1W0, m1b0, m1W1, m1b1, m1W2, m1b2, m1W3, m1b3, pred1);
    k_pairmlp<<<EE, 256, 0, stream>>>(ei, epart, zpart, W1Ts, m2b1, W2Ts, m2b2, m2W3, m2b3, pred1, out);
}
